// deep_Attention_36799279792912
// MI455X (gfx1250) — compile-verified, matrix operand order checked by reading, not run on hardware
//
#include <hip/hip_runtime.h>

typedef __attribute__((ext_vector_type(16))) _Float16 v16h;
typedef __attribute__((ext_vector_type(8)))  float    v8f;
typedef __attribute__((ext_vector_type(4)))  float    f32x4;

#define B_      8
#define C_      64
#define D_      32
#define HW_     4096      // H*W
#define HIDDEN_ 512

// ---------------------------------------------------------------------------
// Kernel 1: fc1 (fp32 dots) -> fc2 (V_WMMA_F32_16X16X32_F16, M=8pad16, N=32,
// K=32 in a single K-step) -> softmax.  One workgroup; writes attn[8][32].
// ---------------------------------------------------------------------------
__global__ __launch_bounds__(256) void attn_softmax_kernel(
    const float* __restrict__ vec, const float* __restrict__ W1,
    const float* __restrict__ b1,  const float* __restrict__ W2,
    const float* __restrict__ b2,  float* __restrict__ attn)
{
    __shared__ float v_lds[B_][D_];   // fc1 output (logit inputs to fc2)
    __shared__ float v2_lds[B_][D_];  // fc2 output
    const int t = threadIdx.x;

    // ---- fc1: one thread per (b,d), full fp32 accuracy ----
    {
        const int b = t >> 5, d = t & 31;
        const float* row = vec + (size_t)(b * D_ + d) * HIDDEN_;
        float acc = 0.f;
        #pragma unroll 8
        for (int h = 0; h < HIDDEN_; ++h) acc = fmaf(row[h], W1[h], acc);
        v_lds[b][d] = acc + b1[0];
    }
    __syncthreads();

    // ---- fc2 via WMMA on wave 0 (EXEC all-1s inside the wave) ----
    // D[m][n] = sum_k A[m][k] * B[k][n], m=batch, n=e, k=d, B[k][n]=W2[n][k].
    if (t < 32) {
        const int lane = t;
        const int r  = lane & 15;   // A row (M) / B column (N within tile)
        const int hi = lane >> 4;   // which K-half this lane holds

        // 16-bit A 16x32 layout (ISA 7.12.2): element j=2p+half holds
        // K = (p>=4 ? 16 : 0) + hi*8 + (p&3)*2 + half, row M = lane&15.
        v16h a;
        #pragma unroll
        for (int j = 0; j < 16; ++j) {
            const int p = j >> 1, half = j & 1;
            const int k = ((p >= 4) ? 16 : 0) + hi * 8 + (p & 3) * 2 + half;
            a[j] = (_Float16)((r < B_) ? v_lds[r][k] : 0.f);
        }

        #pragma unroll
        for (int tile = 0; tile < 2; ++tile) {   // N = 32 -> two 16-wide tiles
            const int n = tile * 16 + r;         // B column-major: lane = N
            v16h bm;
            #pragma unroll
            for (int j = 0; j < 16; ++j) {
                const int p = j >> 1, half = j & 1;
                const int k = ((p >= 4) ? 16 : 0) + hi * 8 + (p & 3) * 2 + half;
                bm[j] = (_Float16)W2[n * D_ + k];   // W2[e][d], e=n, d=k
            }
            v8f c = {};
            c = __builtin_amdgcn_wmma_f32_16x16x32_f16(
                    /*neg_a=*/false, a, /*neg_b=*/false, bm,
                    /*c_mod=*/(short)0, c, /*reuse_a=*/false, /*reuse_b=*/false);
            // D layout: VGPR rr -> M = rr + (lane>=16 ? 8 : 0), N = lane&15.
            #pragma unroll
            for (int rr = 0; rr < 8; ++rr) {
                const int m = rr + hi * 8;
                if (m < B_) v2_lds[m][n] = c[rr] + b2[n];
            }
        }
    }
    __syncthreads();

    // ---- softmax over d (32 values) per batch; trivial, 8 threads ----
    if (t < B_) {
        float mx = v2_lds[t][0];
        #pragma unroll
        for (int d = 1; d < D_; ++d) mx = fmaxf(mx, v2_lds[t][d]);
        float s = 0.f;
        #pragma unroll
        for (int d = 0; d < D_; ++d) s += __expf(v2_lds[t][d] - mx);
        const float inv = 1.f / s;
        #pragma unroll
        for (int d = 0; d < D_; ++d)
            attn[t * D_ + d] = __expf(v2_lds[t][d] - mx) * inv;
    }
}

// ---------------------------------------------------------------------------
// Kernel 2 (the HBM-bound part, ~22us floor at 23.3 TB/s):
// one block per (b,c) plane. Each thread owns 4 float4 positions of the
// 4096-float HW plane: accumulate ctx = sum_d attn[d]*f[...,d,...] with
// streaming (non-temporal) 128-bit loads, then broadcast-store ctx to the
// 32 depth slices with non-temporal 128-bit stores (write-once, no reuse,
// so bypassing cache retention is the right TH policy).
// ---------------------------------------------------------------------------
__global__ __launch_bounds__(256) void ctx_broadcast_kernel(
    const float* __restrict__ f, const float* __restrict__ attn,
    float* __restrict__ out)
{
    const int bc = blockIdx.x;          // b*C + c
    const int b  = bc >> 6;             // C_ = 64
    const f32x4* __restrict__ fp = (const f32x4*)f   + (size_t)bc * (D_ * (HW_ / 4));
    f32x4* __restrict__       op = (f32x4*)out       + (size_t)bc * (D_ * (HW_ / 4));

    __shared__ float a_s[D_];
    if (threadIdx.x < D_) a_s[threadIdx.x] = attn[b * D_ + threadIdx.x];
    __syncthreads();

    const int t = threadIdx.x;
    #pragma unroll
    for (int kblk = 0; kblk < 4; ++kblk) {
        const int pos = t + kblk * 256;          // float4 index in 1024-wide plane
        f32x4 acc = {0.f, 0.f, 0.f, 0.f};
        #pragma unroll
        for (int d = 0; d < D_; ++d) {
            const f32x4 v = __builtin_nontemporal_load(&fp[d * (HW_ / 4) + pos]);
            acc += a_s[d] * v;
        }
        #pragma unroll
        for (int d = 0; d < D_; ++d)
            __builtin_nontemporal_store(acc, &op[d * (HW_ / 4) + pos]);
    }
}

// ---------------------------------------------------------------------------
extern "C" void kernel_launch(void* const* d_in, const int* in_sizes, int n_in,
                              void* d_out, int out_size, void* d_ws, size_t ws_size,
                              hipStream_t stream) {
    const float* vec = (const float*)d_in[0];  // [8,32,512]
    const float* f   = (const float*)d_in[1];  // [8,64,32,64,64]
    const float* W1  = (const float*)d_in[2];  // [1,512]
    const float* b1  = (const float*)d_in[3];  // [1]
    const float* W2  = (const float*)d_in[4];  // [32,32]
    const float* b2  = (const float*)d_in[5];  // [32]
    float* out  = (float*)d_out;
    float* attn = (float*)d_ws;                // 8*32 floats of scratch

    attn_softmax_kernel<<<1, 256, 0, stream>>>(vec, W1, b1, W2, b2, attn);
    ctx_broadcast_kernel<<<B_ * C_, 256, 0, stream>>>(f, attn, out);
}
